// MultiHeadAttnWithRoPE_58669253263442
// MI455X (gfx1250) — matrix operand order repaired and checked by reading, not compiled, not run on hardware
//
#include <hip/hip_runtime.h>
#include <hip/hip_bf16.h>

// MultiHeadAttnWithRoPE for MI455X (gfx1250, wave32, WMMA bf16 path)
//
// Phases:
//  1) weights fp32 -> bf16, TRANSPOSED to [n][k] so B-fragments are contiguous
//  2) Q/K/V = X@W + b (+RoPE for Q,K); Q,K head-major [bh][s][hd],
//     V transposed [bh][hd][s] so the P@V B-fragment is contiguous
//  3) flash attention: async double-buffered K/V chunks in LDS, online softmax
//  4) out = attn @ Wo + bo, fp32 result

typedef __attribute__((ext_vector_type(16))) __bf16 v16bf;
typedef __attribute__((ext_vector_type(8)))  __bf16 v8bf;
typedef __attribute__((ext_vector_type(8)))  float  v8f;
typedef __attribute__((ext_vector_type(4)))  float  v4f;
typedef __attribute__((ext_vector_type(4)))  int    v4i;

#define D_MODEL   1024
#define NUM_HEADS 16
#define HEAD_DIM  64
#define SEQ       2048
#define BATCH     4
#define NTOK      (BATCH * SEQ)   // 8192

#ifndef __has_builtin
#define __has_builtin(x) 0
#endif
#if __has_builtin(__builtin_amdgcn_global_load_async_to_lds_b128)
#define HAVE_ASYNC_BUILTIN 1
#else
#define HAVE_ASYNC_BUILTIN 0
#endif
#if __has_builtin(__builtin_amdgcn_s_wait_asynccnt)
#define HAVE_WAITASYNC_BUILTIN 1
#else
#define HAVE_WAITASYNC_BUILTIN 0
#endif

// ---- gfx1250 async global->LDS copy (ASYNCcnt-tracked) -------------------
typedef __attribute__((address_space(1))) v4i* as1_v4i_ptr;
typedef __attribute__((address_space(3))) v4i* as3_v4i_ptr;

__device__ __forceinline__ void async_copy_b128(const __bf16* g, __bf16* l) {
#if HAVE_ASYNC_BUILTIN
    __builtin_amdgcn_global_load_async_to_lds_b128(
        (as1_v4i_ptr)g, (as3_v4i_ptr)l, 0, 0);
#else
    unsigned loff = (unsigned)(size_t)(__attribute__((address_space(3))) void*)l;
    asm volatile("global_load_async_to_lds_b128 %0, %1, off"
                 :: "v"(loff), "v"(g) : "memory");
#endif
}
__device__ __forceinline__ void wait_async_le(int n /*0 or 16, compile-time*/) {
#if HAVE_WAITASYNC_BUILTIN
    if (n == 0) __builtin_amdgcn_s_wait_asynccnt(0);
    else        __builtin_amdgcn_s_wait_asynccnt(16);
#else
    if (n == 0) asm volatile("s_wait_asynccnt 0x0" ::: "memory");
    else        asm volatile("s_wait_asynccnt 0x10" ::: "memory");
#endif
}

// ---- operand k order used here (wave32, same for A and B) -----------------------------------------
// 16-bit A 16x32: lane L -> row m=L%16; per-lane data = two contiguous
//   8-element K runs starting at hi*16 and hi*16+8 (hi = L>=16).
// 16-bit B 32x16 stored N-major: lane L -> col n=L%16; per-lane data =
//   one contiguous 16-element K run starting at hi*16.
__device__ __forceinline__ v16bf pack_a(v8bf lo, v8bf hi) {
    return __builtin_shufflevector(lo, hi, 0,1,2,3,4,5,6,7,8,9,10,11,12,13,14,15);
}
__device__ __forceinline__ v8f wmma_bf16(v16bf a, v16bf b, v8f c) {
    return __builtin_amdgcn_wmma_f32_16x16x32_bf16(
        false, a, false, b, (short)0, c, false, false);
}

// ---- Phase 1: weight conversion + transpose ------------------------------
// out[n*D + k] = (bf16) in[k*D + n]   (writes coalesced)
__global__ void cvt_w_T_kernel(const float* __restrict__ in,
                               __bf16* __restrict__ out) {
    int o = blockIdx.x * 256 + threadIdx.x;
    int n = o >> 10, k = o & (D_MODEL - 1);
    out[o] = (__bf16)in[(size_t)k * D_MODEL + n];
}

// ---- Phase 2: fused projection (+optional RoPE) --------------------------
// One wave per 16(tokens) x 16(features) tile, K-loop of 32.
// Wt is [n][k] bf16. out: vt==0 -> [bh][s][hd], vt==1 -> [bh][hd][s].
__global__ __launch_bounds__(32)
void qkv_proj_kernel(const float* __restrict__ X,
                     const __bf16* __restrict__ Wt,
                     const float* __restrict__ bias,
                     __bf16* __restrict__ out, int do_rope, int vt)
{
    const int lane = threadIdx.x & 31;
    const int hi   = lane >> 4;
    const int ln   = lane & 15;
    const int tm   = blockIdx.x;
    const int tn   = blockIdx.y;
    const int arow = tm * 16 + ln;
    const int col  = tn * 16 + ln;

    const float* aprow = X + (size_t)arow * D_MODEL;
    const __bf16* bprow = Wt + (size_t)col * D_MODEL;

    v8f acc = {};
    for (int k0 = 0; k0 < D_MODEL; k0 += 32) {
        __builtin_prefetch(aprow + k0 + 128, 0, 1);
        // A fragment: 2x (8 fp32 -> 8 bf16) contiguous runs
        const float* ap = aprow + k0 + hi * 16;
        v4f f0 = *(const v4f*)(ap);
        v4f f1 = *(const v4f*)(ap + 4);
        v4f f2 = *(const v4f*)(ap + 8);
        v4f f3 = *(const v4f*)(ap + 12);
        v16bf a;
#pragma unroll
        for (int i = 0; i < 4; ++i) {
            a[i]      = (__bf16)f0[i];
            a[4 + i]  = (__bf16)f1[i];
            a[8 + i]  = (__bf16)f2[i];
            a[12 + i] = (__bf16)f3[i];
        }
        // B fragment: one contiguous 16-element run (two b128 loads)
        v16bf b = *(const v16bf*)(bprow + k0 + hi * 16);
        acc = wmma_bf16(a, b, acc);
    }

    const float bb = bias[col];
    float c[8];
#pragma unroll
    for (int r = 0; r < 8; ++r) c[r] = acc[r] + bb;

    if (do_rope) {
        // RoPE over the FULL model dim: pair (2i,2i+1), theta = pos*1e4^(-i/512)
        const int   fi   = col >> 1;
        const float freq = __expf(-(float)fi * (9.210340371976184f / 512.0f));
        const bool  even = (col & 1) == 0;
#pragma unroll
        for (int r = 0; r < 8; ++r) {
            int   tok = tm * 16 + r + hi * 8;
            int   pos = tok & (SEQ - 1);
            float ang = (float)pos * freq;
            float sn, cs;
            __sincosf(ang, &sn, &cs);
            float partner = __shfl_xor(c[r], 1, 32);   // col^1 lives in lane^1
            c[r] = even ? (c[r] * cs - partner * sn)
                        : (partner * sn + c[r] * cs);
        }
    }

    const int h  = col >> 6;
    const int dh = col & 63;
#pragma unroll
    for (int r = 0; r < 8; ++r) {
        int tok = tm * 16 + r + hi * 8;
        int bi  = tok >> 11;
        int s   = tok & (SEQ - 1);
        int bh  = bi * NUM_HEADS + h;
        size_t idx = vt ? (((size_t)bh * HEAD_DIM + dh) * SEQ + s)
                        : (((size_t)bh * SEQ + s) * HEAD_DIM + dh);
        out[idx] = (__bf16)c[r];
    }
}

// ---- Phase 3: causal flash attention -------------------------------------
// One wave per (b, h, 16-query tile). 32-key chunks of K ([s][hd]) and
// Vt ([hd][s]) are double-buffered into LDS with async b128 copies.
__global__ __launch_bounds__(32)
void flash_attn_kernel(const __bf16* __restrict__ Q,
                       const __bf16* __restrict__ K,
                       const __bf16* __restrict__ Vt,
                       __bf16* __restrict__ O)
{
    __shared__ __align__(32) __bf16 kbuf[2][32 * HEAD_DIM]; // [key][hd]
    __shared__ __align__(32) __bf16 vbuf[2][HEAD_DIM * 32]; // [hd][key]
    __shared__ __align__(32) __bf16 pbuf[16 * 32];          // P staging

    const int lane = threadIdx.x & 31;
    const int hi   = lane >> 4;
    const int ln   = lane & 15;
    const int qt   = blockIdx.x & (SEQ / 16 - 1);
    const int bh   = blockIdx.x >> 7;
    const int b    = bh >> 4;
    const int h    = bh & (NUM_HEADS - 1);

    const __bf16* Qp  = Q  + (size_t)bh * SEQ * HEAD_DIM;
    const __bf16* Kp  = K  + (size_t)bh * SEQ * HEAD_DIM;
    const __bf16* Vtp = Vt + (size_t)bh * HEAD_DIM * SEQ;

    // Q tile (16x64) in A layout, 2 K-steps
    v16bf qa[2];
#pragma unroll
    for (int ks = 0; ks < 2; ++ks) {
        const __bf16* qp = Qp + (size_t)(qt * 16 + ln) * HEAD_DIM + ks * 32 + hi * 16;
        qa[ks] = pack_a(*(const v8bf*)qp, *(const v8bf*)(qp + 8));
    }

    v8f   oacc[4] = {};
    float rmax[8], rsum[8];
#pragma unroll
    for (int r = 0; r < 8; ++r) { rmax[r] = -1e30f; rsum[r] = 0.0f; }

    const int qhi     = qt * 16 + 15;
    const int nchunks = (qhi >> 5) + 1;

    // prefetch chunk: 8 b128 segments for K, 8 for Vt (256 segs / 32 lanes)
#define ISSUE_CHUNK(kv, kb, vb)                                              \
    do {                                                                     \
        _Pragma("unroll")                                                    \
        for (int i = 0; i < 8; ++i) {                                        \
            int seg = i * 32 + lane;                                         \
            async_copy_b128(Kp + (size_t)((kv) + (seg >> 3)) * HEAD_DIM      \
                                + (seg & 7) * 8,                             \
                            (kb) + seg * 8);                                 \
        }                                                                    \
        _Pragma("unroll")                                                    \
        for (int i = 0; i < 8; ++i) {                                        \
            int seg = i * 32 + lane;                                         \
            async_copy_b128(Vtp + (size_t)(seg >> 2) * SEQ + (kv)            \
                                 + (seg & 3) * 8,                            \
                            (vb) + seg * 8);                                 \
        }                                                                    \
    } while (0)

    ISSUE_CHUNK(0, kbuf[0], vbuf[0]);

    for (int ci = 0; ci < nchunks; ++ci) {
        const int kv  = ci * 32;
        const int cur = ci & 1;
        if (ci + 1 < nchunks) {
            ISSUE_CHUNK(kv + 32, kbuf[cur ^ 1], vbuf[cur ^ 1]);
            wait_async_le(16);      // previous chunk resident, next in flight
        } else {
            wait_async_le(0);
        }
        __syncthreads();
        const __bf16* kb = kbuf[cur];
        const __bf16* vb = vbuf[cur];

        // ---- scores = Q @ K^T (2 N-tiles x 2 K-steps)
        v8f ct[2] = {};
#pragma unroll
        for (int nt = 0; nt < 2; ++nt) {
            const __bf16* krow = kb + (nt * 16 + ln) * HEAD_DIM + hi * 16;
#pragma unroll
            for (int ks = 0; ks < 2; ++ks) {
                v16bf bk = *(const v16bf*)(krow + ks * 32);
                ct[nt] = wmma_bf16(qa[ks], bk, ct[nt]);
            }
        }

        // ---- scale + causal mask
        float sc[2][8];
#pragma unroll
        for (int nt = 0; nt < 2; ++nt)
#pragma unroll
            for (int r = 0; r < 8; ++r) {
                int kglob = kv + nt * 16 + ln;
                int qglob = qt * 16 + r + hi * 8;
                sc[nt][r] = (kglob <= qglob) ? ct[nt][r] * 0.125f : -1e30f;
            }

        // ---- chunk row max (16-lane tree reduce)
        float cmax[8];
#pragma unroll
        for (int r = 0; r < 8; ++r) cmax[r] = fmaxf(sc[0][r], sc[1][r]);
#pragma unroll
        for (int off = 1; off < 16; off <<= 1)
#pragma unroll
            for (int r = 0; r < 8; ++r)
                cmax[r] = fmaxf(cmax[r], __shfl_xor(cmax[r], off, 32));

        // ---- online softmax update
        float corr[8];
#pragma unroll
        for (int r = 0; r < 8; ++r) {
            float nm = fmaxf(rmax[r], cmax[r]);
            corr[r]  = __expf(rmax[r] - nm);
            rmax[r]  = nm;
        }
#pragma unroll
        for (int t = 0; t < 4; ++t)
#pragma unroll
            for (int r = 0; r < 8; ++r) oacc[t][r] *= corr[r];

        float psum[8];
#pragma unroll
        for (int r = 0; r < 8; ++r) {
            float p0 = __expf(sc[0][r] - rmax[r]);
            float p1 = __expf(sc[1][r] - rmax[r]);
            int m = r + hi * 8;
            pbuf[m * 32 + ln]      = (__bf16)p0;
            pbuf[m * 32 + 16 + ln] = (__bf16)p1;
            psum[r] = p0 + p1;
        }
#pragma unroll
        for (int off = 1; off < 16; off <<= 1)
#pragma unroll
            for (int r = 0; r < 8; ++r)
                psum[r] += __shfl_xor(psum[r], off, 32);
#pragma unroll
        for (int r = 0; r < 8; ++r) rsum[r] = rsum[r] * corr[r] + psum[r];

        __syncthreads();

        // ---- P in A layout (16x32)
        const __bf16* pp = pbuf + ln * 32 + hi * 16;
        v16bf pa = pack_a(*(const v8bf*)pp, *(const v8bf*)(pp + 8));

        // ---- O += P @ V (4 N-tiles over hd); Vt chunk is [hd][key]
#pragma unroll
        for (int t = 0; t < 4; ++t) {
            v16bf bv = *(const v16bf*)(vb + (t * 16 + ln) * 32 + hi * 16);
            oacc[t] = wmma_bf16(pa, bv, oacc[t]);
        }
        __syncthreads();
    }
#undef ISSUE_CHUNK

    // ---- normalize, store token-major bf16 [tok][D]
#pragma unroll
    for (int r = 0; r < 8; ++r) {
        float inv = 1.0f / rsum[r];
        int   tok = b * SEQ + qt * 16 + r + hi * 8;
#pragma unroll
        for (int t = 0; t < 4; ++t)
            O[(size_t)tok * D_MODEL + h * HEAD_DIM + t * 16 + ln] =
                (__bf16)(oacc[t][r] * inv);
    }
}

// ---- Phase 4: output projection ------------------------------------------
__global__ __launch_bounds__(32)
void out_proj_kernel(const __bf16* __restrict__ A,     // [NTOK][D] bf16
                     const __bf16* __restrict__ Wt,    // [n][k] bf16
                     const float* __restrict__ bias,
                     float* __restrict__ out)
{
    const int lane = threadIdx.x & 31;
    const int hi   = lane >> 4;
    const int ln   = lane & 15;
    const int arow = blockIdx.x * 16 + ln;
    const int col  = blockIdx.y * 16 + ln;

    const __bf16* aprow = A + (size_t)arow * D_MODEL;
    const __bf16* bprow = Wt + (size_t)col * D_MODEL;

    v8f acc = {};
    for (int k0 = 0; k0 < D_MODEL; k0 += 32) {
        __builtin_prefetch(aprow + k0 + 128, 0, 1);
        const __bf16* ap = aprow + k0 + hi * 16;
        v16bf a = pack_a(*(const v8bf*)ap, *(const v8bf*)(ap + 8));
        v16bf b = *(const v16bf*)(bprow + k0 + hi * 16);
        acc = wmma_bf16(a, b, acc);
    }

    const float bb = bias[col];
#pragma unroll
    for (int r = 0; r < 8; ++r) {
        int tok = blockIdx.x * 16 + r + hi * 8;
        out[(size_t)tok * D_MODEL + col] = acc[r] + bb;
    }
}

// ---- Host launcher --------------------------------------------------------
extern "C" void kernel_launch(void* const* d_in, const int* in_sizes, int n_in,
                              void* d_out, int out_size, void* d_ws, size_t ws_size,
                              hipStream_t stream)
{
    const float* X  = (const float*)d_in[0];
    const float* Wq = (const float*)d_in[1];
    const float* bq = (const float*)d_in[2];
    const float* Wk = (const float*)d_in[3];
    const float* bk = (const float*)d_in[4];
    const float* Wv = (const float*)d_in[5];
    const float* bv = (const float*)d_in[6];
    const float* Wo = (const float*)d_in[7];
    const float* bo = (const float*)d_in[8];
    float* out = (float*)d_out;

    const size_t WELEM = (size_t)D_MODEL * D_MODEL;                   // 1M
    const size_t HELEM = (size_t)BATCH * NUM_HEADS * SEQ * HEAD_DIM;  // 8M
    __bf16* Wq_t = (__bf16*)d_ws;
    __bf16* Wk_t = Wq_t + WELEM;
    __bf16* Wv_t = Wk_t + WELEM;
    __bf16* Wo_t = Wv_t + WELEM;
    __bf16* Qh   = Wo_t + WELEM;   // [bh][s][hd]
    __bf16* Kh   = Qh + HELEM;     // [bh][s][hd]
    __bf16* Vt   = Kh + HELEM;     // [bh][hd][s]
    __bf16* Attn = Vt + HELEM;     // [tok][D]
    (void)ws_size; (void)in_sizes; (void)n_in; (void)out_size;

    {   // 1) weight convert + transpose
        dim3 grid((unsigned)(WELEM / 256)), blk(256);
        cvt_w_T_kernel<<<grid, blk, 0, stream>>>(Wq, Wq_t);
        cvt_w_T_kernel<<<grid, blk, 0, stream>>>(Wk, Wk_t);
        cvt_w_T_kernel<<<grid, blk, 0, stream>>>(Wv, Wv_t);
        cvt_w_T_kernel<<<grid, blk, 0, stream>>>(Wo, Wo_t);
    }
    {   // 2) projections
        dim3 grid(NTOK / 16, D_MODEL / 16), blk(32);
        qkv_proj_kernel<<<grid, blk, 0, stream>>>(X, Wq_t, bq, Qh, 1, 0);
        qkv_proj_kernel<<<grid, blk, 0, stream>>>(X, Wk_t, bk, Kh, 1, 0);
        qkv_proj_kernel<<<grid, blk, 0, stream>>>(X, Wv_t, bv, Vt, 0, 1);
    }
    {   // 3) flash attention
        dim3 grid(BATCH * NUM_HEADS * (SEQ / 16)), blk(32);
        flash_attn_kernel<<<grid, blk, 0, stream>>>(Qh, Kh, Vt, Attn);
    }
    {   // 4) output projection
        dim3 grid(NTOK / 16, D_MODEL / 16), blk(32);
        out_proj_kernel<<<grid, blk, 0, stream>>>(Attn, Wo_t, bo, out);
    }
}
